// LSTM__77034533421656
// MI455X (gfx1250) — matrix operand order repaired and checked by reading, not compiled, not run on hardware
//
#include <hip/hip_runtime.h>

// ---------------------------------------------------------------------------
// Stacked 2-layer LSTM, T=256, B=256, F=128, H=1024 on MI455X (gfx1250).
// Fully fused per-step kernel: bf16 WMMA GEMM for all four gates + in-register
// LSTM cell epilogue. Gate-aware tiling: each wave's 4 accumulators are the
// SAME 16 hidden units across the i/f/g/o blocks, so the nonlinearity needs no
// cross-lane traffic. Recurrent h is double-buffered to avoid the intra-launch
// read/write race; c is updated in place (exclusive ownership per wave).
// Weights (bf16, <= 8MB each) stay resident in the 192MB L2.
// ---------------------------------------------------------------------------

#define T_STEPS 256
#define BATCH   256
#define FDIM    128
#define HDIM    1024
#define GDIM    4096   // 4*H

typedef __attribute__((ext_vector_type(16))) __bf16 v16bf;
typedef __attribute__((ext_vector_type(8)))  float  v8f;

union FragBF { unsigned int u[8]; v16bf v; };

static __device__ __forceinline__ unsigned short f2bf(float f) {
    unsigned int u = __builtin_bit_cast(unsigned int, f);
    u = (u + 0x7FFFu + ((u >> 16) & 1u)) >> 16;   // round-to-nearest-even
    return (unsigned short)u;
}

static __device__ __forceinline__ float fsig(float x) {
    return 1.0f / (1.0f + __expf(-x));
}

// --------------------------- fp32 -> bf16 convert ---------------------------
__global__ __launch_bounds__(256) void cvt_bf16_kernel(
    const float* __restrict__ in, unsigned short* __restrict__ out, int n)
{
    int i = blockIdx.x * 256 + threadIdx.x;
    if (i < n) out[i] = f2bf(in[i]);
}

// --------------------------- fused bias ---------------------------
__global__ __launch_bounds__(256) void bias_fuse_kernel(
    const float* __restrict__ b0, const float* __restrict__ b1,
    float* __restrict__ out, int n)
{
    int i = blockIdx.x * 256 + threadIdx.x;
    if (i < n) out[i] = b0[i] + b1[i];
}

// ---------------------------------------------------------------------------
// Fused LSTM step:
//   gates = A1(B x K1) @ W1(4H x K1)^T + A2(B x K2) @ W2(4H x K2)^T + bias
//   c     = sig(f)*c + sig(i)*tanh(g);   h = sig(o)*tanh(c)
// Grid: 128 blocks x 256 threads (8 waves). Each wave owns a 16-row x
// 16-hidden-unit tile, with 4 accumulators = gates i/f/g/o of those units
// (cols j0, j0+H, j0+2H, j0+3H). 1024 wave-tiles total.
// ---------------------------------------------------------------------------
__global__ __launch_bounds__(256) void lstm_step_kernel(
    const unsigned short* __restrict__ A1, const unsigned short* __restrict__ W1, int K1,
    const unsigned short* __restrict__ A2, const unsigned short* __restrict__ W2, int K2,
    const float* __restrict__ bias,
    float* __restrict__ c,                       // in-place cell state (B x H)
    unsigned short* __restrict__ h_bf_out,       // bf16 h for next step (B x H)
    float* __restrict__ h_f32,                   // optional fp32 h (final step)
    float* __restrict__ h_save)                  // optional fp32 h (t==0, layer 1)
{
    const int lane = threadIdx.x & 31;
    const int wave = threadIdx.x >> 5;
    const int tile = blockIdx.x * 8 + wave;
    const int row0 = (tile >> 6) << 4;   // batch-row tile: (tile / 64) * 16
    const int j0   = (tile & 63) << 4;   // hidden-unit tile: (tile % 64) * 16
    const int hl   = lane >> 4;          // lane half (K-half select)
    const int mn   = lane & 15;          // row (A) / col (B,C) within tile

    v8f acc[4];                          // acc[g] = gate g (i,f,g,o) partials
#pragma unroll
    for (int g = 0; g < 4; ++g)
#pragma unroll
        for (int v = 0; v < 8; ++v) acc[g][v] = 0.0f;

#pragma unroll 1
    for (int seg = 0; seg < 2; ++seg) {
        const unsigned short* A = seg ? A2 : A1;
        const unsigned short* W = seg ? W2 : W1;
        const int K = seg ? K2 : K1;

        const unsigned int* Arow = (const unsigned int*)(A + (long)(row0 + mn) * K);

        for (int k0 = 0; k0 < K; k0 += 32) {
            // A fragment: 16x32 bf16 (lane<16 K={0..7,16..23}, lane>=16
            // K={8..15,24..31}, dword pairs per the ISA VGPR layout)
            FragBF a;
            const unsigned int* ap = Arow + (k0 >> 1);
#pragma unroll
            for (int v = 0; v < 8; ++v)
                a.u[v] = ap[v + 4 * hl + ((v >= 4) ? 4 : 0)];

            // 4 B fragments: rows (j0+mn) of W within each gate block
#pragma unroll
            for (int g = 0; g < 4; ++g) {
                const unsigned int* wp =
                    (const unsigned int*)(W + (long)(g * HDIM + j0 + mn) * K)
                    + (k0 >> 1) + 4 * hl;
                FragBF b;
#pragma unroll
                for (int v = 0; v < 8; ++v) b.u[v] = wp[v + ((v >= 4) ? 4 : 0)];

                acc[g] = __builtin_amdgcn_wmma_f32_16x16x32_bf16(
                    false, a.v, false, b.v, (short)0, acc[g], false, false);
            }
        }
    }

    // ---- fused cell epilogue: lane holds i/f/g/o for 8 (row, j) elements ----
    const int j = j0 + mn;
    const float bi = bias[j];
    const float bf_ = bias[HDIM + j];
    const float bg = bias[2 * HDIM + j];
    const float bo = bias[3 * HDIM + j];

#pragma unroll
    for (int v = 0; v < 8; ++v) {
        const int row = row0 + v + 8 * hl;       // C/D layout: c[v]=C[v+8*hl][mn]
        const long idx = (long)row * HDIM + j;

        const float ig = fsig(acc[0][v] + bi);
        const float fg = fsig(acc[1][v] + bf_);
        const float gg = tanhf(acc[2][v] + bg);
        const float og = fsig(acc[3][v] + bo);

        const float cn = fg * c[idx] + ig * gg;
        c[idx] = cn;
        const float hn = og * tanhf(cn);
        h_bf_out[idx] = f2bf(hn);
        if (h_f32)  h_f32[idx]  = hn;
        if (h_save) h_save[idx] = hn;
    }
}

// --------------------------- final linear: y = h1(t=0) @ W_lin^T + b ---------
__global__ __launch_bounds__(256) void linear_kernel(
    const float* __restrict__ h, const float* __restrict__ W,
    const float* __restrict__ b, float* __restrict__ y)
{
    const int idx = blockIdx.x * 256 + threadIdx.x;   // < B*F
    const int bb = idx >> 7;
    const int f  = idx & 127;
    const float* hr = h + (long)bb * HDIM;
    const float* wr = W + (long)f * HDIM;
    float s = 0.0f;
#pragma unroll 4
    for (int j = 0; j < HDIM; ++j) s = fmaf(hr[j], wr[j], s);
    y[idx] = s + b[f];
}

// --------------------------- pack h_n / c_n into d_out ---------------------------
__global__ __launch_bounds__(256) void pack_kernel(
    const float* __restrict__ h0, const float* __restrict__ h1,
    const float* __restrict__ c0, const float* __restrict__ c1,
    float* __restrict__ out)   // out points at d_out + B*F
{
    const int idx = blockIdx.x * 256 + threadIdx.x;   // < 2*B*H
    const int layer = idx >> 18;                      // / (B*H)
    const int off   = idx & (BATCH * HDIM - 1);
    out[idx]                     = (layer ? h1 : h0)[off];
    out[2 * BATCH * HDIM + idx]  = (layer ? c1 : c0)[off];
}

// ---------------------------------------------------------------------------
extern "C" void kernel_launch(void* const* d_in, const int* in_sizes, int n_in,
                              void* d_out, int out_size, void* d_ws, size_t ws_size,
                              hipStream_t stream) {
    (void)in_sizes; (void)n_in; (void)out_size; (void)ws_size;

    const float* x     = (const float*)d_in[0];
    const float* W_ih0 = (const float*)d_in[1];
    const float* W_hh0 = (const float*)d_in[2];
    const float* b_ih0 = (const float*)d_in[3];
    const float* b_hh0 = (const float*)d_in[4];
    const float* W_ih1 = (const float*)d_in[5];
    const float* W_hh1 = (const float*)d_in[6];
    const float* b_ih1 = (const float*)d_in[7];
    const float* b_hh1 = (const float*)d_in[8];
    const float* W_lin = (const float*)d_in[9];
    const float* b_lin = (const float*)d_in[10];

    float* out = (float*)d_out;

    // ---- workspace carve-up (bytes) ----
    char* ws = (char*)d_ws;
    size_t off = 0;
    auto carve = [&](size_t bytes) { char* p = ws + off; off += (bytes + 255) & ~(size_t)255; return p; };

    unsigned short* x_bf    = (unsigned short*)carve((size_t)T_STEPS * BATCH * FDIM * 2); // 16 MB
    unsigned short* Wih0_bf = (unsigned short*)carve((size_t)GDIM * FDIM * 2);            // 1 MB
    unsigned short* Whh0_bf = (unsigned short*)carve((size_t)GDIM * HDIM * 2);            // 8 MB
    unsigned short* Wih1_bf = (unsigned short*)carve((size_t)GDIM * HDIM * 2);            // 8 MB
    unsigned short* Whh1_bf = (unsigned short*)carve((size_t)GDIM * HDIM * 2);            // 8 MB
    float* bias0  = (float*)carve((size_t)GDIM * 4);
    float* bias1  = (float*)carve((size_t)GDIM * 4);
    float* c0     = (float*)carve((size_t)BATCH * HDIM * 4);
    float* c1     = (float*)carve((size_t)BATCH * HDIM * 4);
    float* h0f    = (float*)carve((size_t)BATCH * HDIM * 4);
    float* h1f    = (float*)carve((size_t)BATCH * HDIM * 4);
    unsigned short* h0b[2], * h1b[2];
    h0b[0] = (unsigned short*)carve((size_t)BATCH * HDIM * 2);
    h0b[1] = (unsigned short*)carve((size_t)BATCH * HDIM * 2);
    h1b[0] = (unsigned short*)carve((size_t)BATCH * HDIM * 2);
    h1b[1] = (unsigned short*)carve((size_t)BATCH * HDIM * 2);
    float* h1t0   = (float*)carve((size_t)BATCH * HDIM * 4);

    // ---- zero initial state (h "prev" buffer for t=0 is index 1) ----
    hipMemsetAsync(c0,     0, (size_t)BATCH * HDIM * 4, stream);
    hipMemsetAsync(c1,     0, (size_t)BATCH * HDIM * 4, stream);
    hipMemsetAsync(h0b[1], 0, (size_t)BATCH * HDIM * 2, stream);
    hipMemsetAsync(h1b[1], 0, (size_t)BATCH * HDIM * 2, stream);

    // ---- convert to bf16, fuse biases ----
    auto cvt = [&](const float* in, unsigned short* o, int n) {
        cvt_bf16_kernel<<<(n + 255) / 256, 256, 0, stream>>>(in, o, n);
    };
    cvt(x,     x_bf,    T_STEPS * BATCH * FDIM);
    cvt(W_ih0, Wih0_bf, GDIM * FDIM);
    cvt(W_hh0, Whh0_bf, GDIM * HDIM);
    cvt(W_ih1, Wih1_bf, GDIM * HDIM);
    cvt(W_hh1, Whh1_bf, GDIM * HDIM);
    bias_fuse_kernel<<<(GDIM + 255) / 256, 256, 0, stream>>>(b_ih0, b_hh0, bias0, GDIM);
    bias_fuse_kernel<<<(GDIM + 255) / 256, 256, 0, stream>>>(b_ih1, b_hh1, bias1, GDIM);

    const int stepGrid = (BATCH / 16) * (HDIM / 16) / 8;   // 1024 wave-tiles / 8 = 128

    // ---- 256 serial timesteps; stream order provides all dependencies ----
    for (int t = 0; t < T_STEPS; ++t) {
        const int cur = t & 1, prev = cur ^ 1;
        const bool last = (t == T_STEPS - 1);

        // layer 0: h0[cur] = cell(x_t @ Wih0^T + h0[prev] @ Whh0^T + bias0)
        lstm_step_kernel<<<stepGrid, 256, 0, stream>>>(
            x_bf + (size_t)t * BATCH * FDIM, Wih0_bf, FDIM,
            h0b[prev], Whh0_bf, HDIM, bias0,
            c0, h0b[cur], last ? h0f : nullptr, nullptr);

        // layer 1: h1[cur] = cell(h0[cur] @ Wih1^T + h1[prev] @ Whh1^T + bias1)
        lstm_step_kernel<<<stepGrid, 256, 0, stream>>>(
            h0b[cur], Wih1_bf, HDIM,
            h1b[prev], Whh1_bf, HDIM, bias1,
            c1, h1b[cur], last ? h1f : nullptr, (t == 0) ? h1t0 : nullptr);
    }

    // ---- y = out1[0] @ W_lin^T + b_lin ----
    linear_kernel<<<(BATCH * FDIM) / 256, 256, 0, stream>>>(h1t0, W_lin, b_lin, out);

    // ---- h_n / c_n ----
    pack_kernel<<<(2 * BATCH * HDIM) / 256, 256, 0, stream>>>(
        h0f, h1f, c0, c1, out + BATCH * FDIM);
}
